// CustomGCNLayer_81252191306415
// MI455X (gfx1250) — matrix operand order repaired and checked by reading, not compiled, not run on hardware
//
#include <hip/hip_runtime.h>
#include <hip/hip_bf16.h>

// CDNA5 / gfx1250 GCN layer:
//   out = relu( D^-1/2 (A+I) D^-1/2 (xW + b) )
// Heavy op: dense A (10000x10000 fp32, 0/1 values) @ hT  via V_WMMA_F32_16X16X32_F16.
// 0/1 adjacency is exact in f16; B-side (d * h) is f16 with fp32 accumulation.
// Self-loop folded into the epilogue: out_i = relu(d_i * (acc_i + hT[:,i])).

typedef __attribute__((ext_vector_type(16))) _Float16 v16h;
typedef __attribute__((ext_vector_type(8)))  float    v8f;

#define DINOUT 64

// ---------------------------------------------------------------------------
// Kernel 1: d_is[i] = rsqrt(1 + sum_j adj[i][j])   (self loop => degree >= 1)
// One 256-thread block per row, coalesced strided reads, LDS tree reduce.
// ---------------------------------------------------------------------------
__global__ void gcn_degree_kernel(const float* __restrict__ adj,
                                  float* __restrict__ d_is, int n) {
  const int row = blockIdx.x;
  const int tid = threadIdx.x;
  __shared__ float red[256];
  const float* arow = adj + (size_t)row * (size_t)n;
  float s = 0.0f;
  for (int c = tid; c < n; c += 256) s += arow[c];
  red[tid] = s;
  __syncthreads();
  for (int off = 128; off > 0; off >>= 1) {
    if (tid < off) red[tid] += red[tid + off];
    __syncthreads();
  }
  if (tid == 0) d_is[row] = rsqrtf(1.0f + red[0]);
}

// ---------------------------------------------------------------------------
// Kernel 2: hT[col][row] = f16( d_is[row] * (x[row,:]@W[:,col] + b[col]) )
// Stored TRANSPOSED [64][kpad] so WMMA B-fragments are contiguous 32B per lane.
// Rows n..kpad-1 zero-filled (K padding for the 16x16x32 WMMA chunks).
// ---------------------------------------------------------------------------
__global__ void gcn_linear_kernel(const float* __restrict__ x,
                                  const float* __restrict__ W,
                                  const float* __restrict__ bias,
                                  const float* __restrict__ d_is,
                                  _Float16* __restrict__ hT,
                                  int n, int kpad) {
  const int row = blockIdx.x;     // 0..kpad-1
  const int col = threadIdx.x;    // 0..63
  __shared__ float xs[DINOUT];
  if (row < n) xs[col] = x[(size_t)row * DINOUT + col];
  __syncthreads();
  if (row >= n) {                 // zero-fill K padding
    hT[(size_t)col * kpad + row] = (_Float16)0.0f;
    return;
  }
  float s = bias[col];
#pragma unroll
  for (int k = 0; k < DINOUT; ++k) s = fmaf(xs[k], W[k * DINOUT + col], s);
  hT[(size_t)col * kpad + row] = (_Float16)(d_is[row] * s);
}

// ---------------------------------------------------------------------------
// Kernel 3: out = relu( d_i * (sum_j adj[i][j] * hT[:,j] + hT[:,i]) )
// 128 threads = 4 waves per block; each wave computes one 16-row strip x 64 cols
// as 4 accumulators of V_WMMA_F32_16X16X32_F16 over K chunks of 32.
//
// A-frag 16x32 f16 layout: lanes 0-15 row=lane, K rel 0..15;
// lanes 16-31 row=lane-16, K rel 16..31  -> four b128 loads + cvt.
// B-frag 32x16 f16: lanes 0-15 col=lane K rel 0..15; lanes 16-31 col=lane-16
// K rel 16..31 -> one contiguous 32B v16h load per N-tile from transposed hT.
// ---------------------------------------------------------------------------
__global__ void __launch_bounds__(128, 1)
gcn_wmma_kernel(const float* __restrict__ adj,
                const _Float16* __restrict__ hT,
                const float* __restrict__ d_is,
                float* __restrict__ out,
                int n, int kpad) {
  const int wave = threadIdx.x >> 5;
  const int strip = blockIdx.x * 4 + wave;
  const int rowBase = strip * 16;
  if (rowBase >= n) return;       // n % 16 == 0 for this problem

  const int lane = threadIdx.x & 31;
  const int half = lane >> 4;     // 0: lanes 0-15, 1: lanes 16-31
  const int lidx = lane & 15;

  const float* arow = adj + (size_t)(rowBase + lidx) * (size_t)n;
  const _Float16* brow = hT + (size_t)lidx * (size_t)kpad + half * 16;

  v8f acc0 = {}, acc1 = {}, acc2 = {}, acc3 = {};

  const int kchunks = (n + 31) >> 5;
  for (int kc = 0; kc < kchunks; ++kc) {
    const int k0 = kc << 5;

    // ---- A fragment: 16 adjacency floats for this lane -> f16 ----
    const float* ap = arow + k0 + half * 16;
    float4 f0, f1, f2, f3;
    if (k0 + half * 16 + 16 <= n) {               // only final chunk is partial
      f0 = *(const float4*)(ap + 0);              // K rel +0..3  (or +16..19)
      f1 = *(const float4*)(ap + 4);              // K rel +4..7  (or +20..23)
      f2 = *(const float4*)(ap + 8);              // K rel +8..11 (or +24..27)
      f3 = *(const float4*)(ap + 12);             // K rel +12..15 (or +28..31)
    } else {                                      // n % 32 == 16: tail K is OOB
      f0 = make_float4(0.f, 0.f, 0.f, 0.f);
      f1 = make_float4(0.f, 0.f, 0.f, 0.f);
      f2 = make_float4(0.f, 0.f, 0.f, 0.f);
      f3 = make_float4(0.f, 0.f, 0.f, 0.f);
    }
    v16h a;
    a[0]  = (_Float16)f0.x; a[1]  = (_Float16)f0.y;
    a[2]  = (_Float16)f0.z; a[3]  = (_Float16)f0.w;
    a[4]  = (_Float16)f1.x; a[5]  = (_Float16)f1.y;
    a[6]  = (_Float16)f1.z; a[7]  = (_Float16)f1.w;
    a[8]  = (_Float16)f2.x; a[9]  = (_Float16)f2.y;
    a[10] = (_Float16)f2.z; a[11] = (_Float16)f2.w;
    a[12] = (_Float16)f3.x; a[13] = (_Float16)f3.y;
    a[14] = (_Float16)f3.z; a[15] = (_Float16)f3.w;

    // ---- B fragments: 32 contiguous bytes per lane from transposed hT ----
    const _Float16* bp = brow + k0;
    const v16h b0 = *(const v16h*)(bp);                       // cols  0..15
    const v16h b1 = *(const v16h*)(bp + 16 * (size_t)kpad);   // cols 16..31
    const v16h b2 = *(const v16h*)(bp + 32 * (size_t)kpad);   // cols 32..47
    const v16h b3 = *(const v16h*)(bp + 48 * (size_t)kpad);   // cols 48..63

    acc0 = __builtin_amdgcn_wmma_f32_16x16x32_f16(false, a, false, b0,
                                                  (short)0, acc0, false, false);
    acc1 = __builtin_amdgcn_wmma_f32_16x16x32_f16(false, a, false, b1,
                                                  (short)0, acc1, false, false);
    acc2 = __builtin_amdgcn_wmma_f32_16x16x32_f16(false, a, false, b2,
                                                  (short)0, acc2, false, false);
    acc3 = __builtin_amdgcn_wmma_f32_16x16x32_f16(false, a, false, b3,
                                                  (short)0, acc3, false, false);
  }

  // ---- Epilogue: self-loop term + D^-1/2 scale + ReLU ----
  // C/D layout: VGPR r -> M = r (lanes 0-15) / r+8 (lanes 16-31), N = lane&15.
#pragma unroll
  for (int r = 0; r < 8; ++r) {
    const int orow = rowBase + r + half * 8;
    const float d = d_is[orow];
    float* orow_ptr = out + (size_t)orow * DINOUT;
#pragma unroll
    for (int nt = 0; nt < 4; ++nt) {
      const int col = nt * 16 + lidx;
      const float accv = (nt == 0) ? acc0[r]
                       : (nt == 1) ? acc1[r]
                       : (nt == 2) ? acc2[r]
                                   : acc3[r];
      const float selfv = (float)hT[(size_t)col * kpad + orow]; // = d_row*h[row,col]
      orow_ptr[col] = fmaxf(d * (accv + selfv), 0.0f);
    }
  }
}

// ---------------------------------------------------------------------------
// Launch
// ---------------------------------------------------------------------------
extern "C" void kernel_launch(void* const* d_in, const int* in_sizes, int n_in,
                              void* d_out, int out_size, void* d_ws, size_t ws_size,
                              hipStream_t stream) {
  const float* x    = (const float*)d_in[0];   // [n, 64]
  const float* adj  = (const float*)d_in[1];   // [n, n]
  const float* W    = (const float*)d_in[2];   // [64, 64]
  const float* bias = (const float*)d_in[3];   // [64]
  float* out        = (float*)d_out;           // [n, 64]

  const int n    = in_sizes[0] / DINOUT;       // 10000
  const int kpad = (n + 31) & ~31;             // 10016 (K padding for WMMA)

  // Workspace layout: [0, 64KB): d_is (n floats); [64KB, ...): hT f16[64][kpad]
  float*    d_is = (float*)d_ws;
  _Float16* hT   = (_Float16*)((char*)d_ws + (64 << 10));

  gcn_degree_kernel<<<n, 256, 0, stream>>>(adj, d_is, n);
  gcn_linear_kernel<<<kpad, DINOUT, 0, stream>>>(x, W, bias, d_is, hT, n, kpad);

  const int strips = n / 16;                    // 625
  const int blocks = (strips + 3) / 4;          // 4 waves (strips) per block
  gcn_wmma_kernel<<<blocks, 128, 0, stream>>>(adj, hT, d_is, out, n, kpad);
}
